// TransformerNet_24807731102209
// MI455X (gfx1250) — matrix operand order repaired and checked by reading, not compiled, not run on hardware
//
#include <hip/hip_runtime.h>
#include <hip/hip_bf16.h>

typedef __attribute__((ext_vector_type(16))) __bf16   v16bf;
typedef __attribute__((ext_vector_type(8)))  float    v8f;
typedef __attribute__((ext_vector_type(8)))  unsigned v8u;

#define N_NODES  61184
#define N_EDGES  128000
#define N_GRAPHS 128

// ---------- helpers ----------
__device__ __forceinline__ unsigned short f2bf(float f) {
  unsigned u = __float_as_uint(f);
  unsigned r = u + 0x7FFFu + ((u >> 16) & 1u);  // round-to-nearest-even
  return (unsigned short)(r >> 16);
}
// monotonic float<->uint mapping for atomicMax-based segment max
__device__ __forceinline__ unsigned encf(float f) {
  unsigned u = __float_as_uint(f);
  return (u & 0x80000000u) ? ~u : (u | 0x80000000u);
}
__device__ __forceinline__ float decf(unsigned k) {
  unsigned u = (k & 0x80000000u) ? (k & 0x7FFFFFFFu) : ~k;
  return __uint_as_float(u);
}

// ---------- generic zero ----------
__global__ void k_zero(unsigned* p, long long n) {
  long long i = (long long)blockIdx.x * blockDim.x + threadIdx.x;
  if (i < n) p[i] = 0u;
}

// ---------- cast x (N x 3) into bf16 padded to K=32 ----------
__global__ void k_cast_x(const float* __restrict__ x, unsigned short* __restrict__ hb) {
  int i = blockIdx.x * blockDim.x + threadIdx.x;
  if (i >= N_NODES * 32) return;
  int n = i >> 5, k = i & 31;
  hb[i] = (k < 3) ? f2bf(x[n * 3 + k]) : (unsigned short)0;
}

// ---------- pack [Wq|Wk|Wv|Ws] -> bf16 Kpad x M (row-major), bias concat ----------
__global__ void k_pack_w(const float* __restrict__ Wq, const float* __restrict__ Wk,
                         const float* __restrict__ Wv, const float* __restrict__ Ws,
                         const float* __restrict__ bq, const float* __restrict__ bk,
                         const float* __restrict__ bv, const float* __restrict__ bs,
                         unsigned short* __restrict__ Wc, float* __restrict__ bc,
                         int fi, int fo, int Kpad) {
  int M = 4 * fo;
  int i = blockIdx.x * blockDim.x + threadIdx.x;
  if (i >= Kpad * M) return;
  int k = i / M, m = i % M;
  int g = m / fo, mc = m % fo;
  const float* W = (g == 0) ? Wq : (g == 1) ? Wk : (g == 2) ? Wv : Ws;
  float v = (k < fi) ? W[k * fo + mc] : 0.f;
  Wc[i] = f2bf(v);
  if (k == 0) {
    const float* b = (g == 0) ? bq : (g == 1) ? bk : (g == 2) ? bv : bs;
    bc[m] = b[mc];
  }
}

// ---------- WMMA GEMM: C[N x M] = A[N x Kpad](bf16) * B[Kpad x M](bf16) + bias ----------
// Register-blocked: one wave computes a 32x64 C macro-tile = 2x4 grid of 16x16 WMMA
// tiles (8 accumulators, ~145 VGPRs total -> no spills).
// Per 32-wide K-step: 2 A frags + 4 B frags -> 8 WMMAs.
// VGPR layouts per CDNA5 ISA 7.12.2:
//   A: lane%16 = row, half=lane/16; vgpr v holds K pair (half*8+2v) / (16+half*8+2(v-4))
//   B: lane%16 = column, half=lane/16; vgpr v holds K pair as A  (row-major K x M)
__global__ void k_gemm(const unsigned short* __restrict__ A,
                       const unsigned short* __restrict__ B,
                       const float* __restrict__ bias, float* __restrict__ C,
                       int Kpad, int M, int tiles_m64, int tiles_r32) {
  int wave = (int)(((long long)blockIdx.x * blockDim.x + threadIdx.x) >> 5);
  int lane = threadIdx.x & 31;
  int tr = wave / tiles_m64, tc = wave % tiles_m64;
  if (tr >= tiles_r32) return;
  int l15 = lane & 15, half = lane >> 4;

  const unsigned short* arow0 = A + (size_t)(tr * 32 + l15) * Kpad;
  const unsigned short* bbase = B + (size_t)tc * 64;

  v8f acc[2][4] = {};
  for (int kb = 0; kb < Kpad; kb += 32) {
    // A fragments (2 row tiles); row tile stride = 16 rows = 16*Kpad shorts
    v16bf afrag[2];
#pragma unroll
    for (int i = 0; i < 2; ++i) {
      const unsigned short* ap = arow0 + (size_t)i * 16 * Kpad + kb;
      v8u au;
#pragma unroll
      for (int v = 0; v < 8; ++v) {
        int k0 = (v < 4) ? (half * 8 + 2 * v) : (16 + half * 8 + 2 * (v - 4));
        au[v] = *(const unsigned*)(ap + k0);
      }
      afrag[i] = __builtin_bit_cast(v16bf, au);
    }
    // B fragments (4 col tiles); lane's column of B, K strided by M
    const unsigned short* bp = bbase + (size_t)kb * M + l15;
    v16bf bfrag[4];
#pragma unroll
    for (int j = 0; j < 4; ++j) {
      v8u bu;
#pragma unroll
      for (int v = 0; v < 8; ++v) {
        int k0 = (v < 4) ? (half * 8 + 2 * v) : (16 + half * 8 + 2 * (v - 4));
        bu[v] = (unsigned)bp[(size_t)k0 * M + j * 16] |
                ((unsigned)bp[(size_t)(k0 + 1) * M + j * 16] << 16);
      }
      bfrag[j] = __builtin_bit_cast(v16bf, bu);
    }
    // prefetch next K-block of B (global_prefetch_b8)
    if (kb + 32 < Kpad)
      __builtin_prefetch(bp + (size_t)32 * M, 0, 1);
    // 8 WMMAs per K-step
#pragma unroll
    for (int i = 0; i < 2; ++i)
#pragma unroll
      for (int j = 0; j < 4; ++j)
        acc[i][j] = __builtin_amdgcn_wmma_f32_16x16x32_bf16(
            false, afrag[i], false, bfrag[j], (short)0, acc[i][j], false, false);
  }

  // epilogue: bias add + f32 store
#pragma unroll
  for (int j = 0; j < 4; ++j) {
    int col = tc * 64 + j * 16 + l15;
    float bv = bias[col];
#pragma unroll
    for (int i = 0; i < 2; ++i) {
#pragma unroll
      for (int v = 0; v < 8; ++v) {
        int row = tr * 32 + i * 16 + half * 8 + v;
        C[(size_t)row * M + col] = acc[i][j][v] + bv;
      }
    }
  }
}

// ---------- per-edge attention score + segment max ----------
__global__ void k_scores(const float* __restrict__ qkvs, const int* __restrict__ src,
                         const int* __restrict__ dst, float* __restrict__ scores,
                         unsigned* __restrict__ mseg, int fo) {
  int e = (int)(((long long)blockIdx.x * blockDim.x + threadIdx.x) >> 5);
  int lane = threadIdx.x & 31;
  if (e >= N_EDGES) return;
  int ld = 4 * fo;
  int d = dst[e];
  const float* q = qkvs + (size_t)d * ld;           // q columns [0, fo)
  const float* k = qkvs + (size_t)src[e] * ld + fo; // k columns [fo, 2fo)
  float s = 0.f;
  for (int c = lane; c < fo; c += 32) s += q[c] * k[c];
  for (int off = 16; off > 0; off >>= 1) s += __shfl_xor(s, off, 32);
  if (lane == 0) {
    s *= rsqrtf((float)fo);
    scores[e] = s;
    atomicMax(&mseg[d], encf(s));
  }
}

// ---------- exp(score - max) + segment sum ----------
__global__ void k_expden(const float* __restrict__ scores, const int* __restrict__ dst,
                         const unsigned* __restrict__ mseg, float* __restrict__ aexp,
                         float* __restrict__ den) {
  int e = blockIdx.x * blockDim.x + threadIdx.x;
  if (e >= N_EDGES) return;
  int d = dst[e];
  float a = __expf(scores[e] - decf(mseg[d]));
  aexp[e] = a;
  atomicAdd(&den[d], a);
}

// ---------- weighted scatter of v into agg ----------
__global__ void k_agg(const float* __restrict__ qkvs, const int* __restrict__ src,
                      const int* __restrict__ dst, const float* __restrict__ aexp,
                      const float* __restrict__ den, float* __restrict__ agg, int fo) {
  long long i = (long long)blockIdx.x * blockDim.x + threadIdx.x;
  if (i >= (long long)N_EDGES * fo) return;
  int e = (int)(i / fo), c = (int)(i % fo);
  int d = dst[e];
  float a = aexp[e] / den[d];
  float vv = qkvs[(size_t)src[e] * 4 * fo + 2 * fo + c]; // v columns [2fo, 3fo)
  atomicAdd(&agg[(size_t)d * fo + c], a * vv);
}

// ---------- h = ELU(agg + skip); also emit bf16 for next layer's GEMM ----------
__global__ void k_combine(const float* __restrict__ agg, const float* __restrict__ qkvs,
                          float* __restrict__ h, unsigned short* __restrict__ hb, int fo) {
  long long i = (long long)blockIdx.x * blockDim.x + threadIdx.x;
  if (i >= (long long)N_NODES * fo) return;
  int n = (int)(i / fo), c = (int)(i % fo);
  float x = agg[i] + qkvs[(size_t)n * 4 * fo + 3 * fo + c]; // s columns [3fo, 4fo)
  float y = (x > 0.f) ? x : expm1f(x);
  h[i] = y;
  hb[i] = f2bf(y);
}

// ---------- gate = h @ Wg + bg ; segment max over graphs ----------
__global__ void k_gate(const float* __restrict__ h, const float* __restrict__ Wg,
                       const float* __restrict__ bg, const int* __restrict__ batch,
                       float* __restrict__ gate, unsigned* __restrict__ gm) {
  int n = (int)(((long long)blockIdx.x * blockDim.x + threadIdx.x) >> 5);
  int lane = threadIdx.x & 31;
  if (n >= N_NODES) return;
  float s = h[n * 32 + lane] * Wg[lane];
  for (int off = 16; off > 0; off >>= 1) s += __shfl_xor(s, off, 32);
  if (lane == 0) {
    s += bg[0];
    gate[n] = s;
    atomicMax(&gm[batch[n]], encf(s));
  }
}

__global__ void k_poolexp(float* __restrict__ gate, const int* __restrict__ batch,
                          const unsigned* __restrict__ gm, float* __restrict__ gden) {
  int n = blockIdx.x * blockDim.x + threadIdx.x;
  if (n >= N_NODES) return;
  int g = batch[n];
  float e = __expf(gate[n] - decf(gm[g]));
  gate[n] = e;
  atomicAdd(&gden[g], e);
}

__global__ void k_pool(const float* __restrict__ h, const float* __restrict__ gate,
                       const int* __restrict__ batch, const float* __restrict__ gden,
                       float* __restrict__ pooled) {
  int i = blockIdx.x * blockDim.x + threadIdx.x;
  if (i >= N_NODES * 32) return;
  int n = i >> 5, c = i & 31;
  int g = batch[n];
  atomicAdd(&pooled[g * 32 + c], (gate[n] / gden[g]) * h[i]);
}

__global__ void k_final(const float* __restrict__ pooled, const float* __restrict__ Wfc,
                        const float* __restrict__ bfc, float* __restrict__ out) {
  int i = blockIdx.x * blockDim.x + threadIdx.x;
  if (i >= N_GRAPHS * 3) return;
  int g = i / 3, j = i % 3;
  float s = bfc[j];
#pragma unroll
  for (int c = 0; c < 32; ++c) s += pooled[g * 32 + c] * Wfc[c * 3 + j];
  out[i] = s;
}

// ---------- host ----------
extern "C" void kernel_launch(void* const* d_in, const int* in_sizes, int n_in,
                              void* d_out, int out_size, void* d_ws, size_t ws_size,
                              hipStream_t stream) {
  const int N = N_NODES, E = N_EDGES, G = N_GRAPHS;
  const float* x     = (const float*)d_in[0];
  const int*   src   = (const int*)d_in[1];
  const int*   dst   = src + E;
  const int*   batch = (const int*)d_in[2];
  const float* Wg    = (const float*)d_in[35];
  const float* bg    = (const float*)d_in[36];
  const float* Wfc   = (const float*)d_in[37];
  const float* bfc   = (const float*)d_in[38];

  size_t off = 0;
  auto alloc = [&](size_t bytes) -> void* {
    char* p = (char*)d_ws + off;
    off += (bytes + 255) & ~(size_t)255;
    return (void*)p;
  };
  float*          h      = (float*)alloc((size_t)N * 512 * 4);
  unsigned short* hb     = (unsigned short*)alloc((size_t)N * 512 * 2);
  float*          qkvs   = (float*)alloc((size_t)N * 2048 * 4);
  float*          agg    = (float*)alloc((size_t)N * 512 * 4);
  unsigned short* Wc     = (unsigned short*)alloc((size_t)512 * 2048 * 2);
  float*          bc     = (float*)alloc(2048 * 4);
  float*          scores = (float*)alloc((size_t)E * 4);
  float*          aexp   = (float*)alloc((size_t)E * 4);
  unsigned*       mseg   = (unsigned*)alloc((size_t)N * 4);
  float*          den    = (float*)alloc((size_t)N * 4);
  float*          gate   = (float*)alloc((size_t)N * 4);
  unsigned*       gm     = (unsigned*)alloc(G * 4);
  float*          gden   = (float*)alloc(G * 4);
  float*          pooled = (float*)alloc(G * 32 * 4);

  const int FI[4] = {3, 512, 256, 64};
  const int FO[4] = {512, 256, 64, 32};

  k_cast_x<<<(N * 32 + 255) / 256, 256, 0, stream>>>(x, hb);

  for (int l = 0; l < 4; ++l) {
    int fi = FI[l], fo = FO[l], M = 4 * fo;
    int Kpad = (fi + 31) & ~31;
    const float* Wq = (const float*)d_in[3 + 8 * l + 0];
    const float* bq = (const float*)d_in[3 + 8 * l + 1];
    const float* Wk = (const float*)d_in[3 + 8 * l + 2];
    const float* bk = (const float*)d_in[3 + 8 * l + 3];
    const float* Wv = (const float*)d_in[3 + 8 * l + 4];
    const float* bv = (const float*)d_in[3 + 8 * l + 5];
    const float* Ws = (const float*)d_in[3 + 8 * l + 6];
    const float* bs = (const float*)d_in[3 + 8 * l + 7];

    k_pack_w<<<(Kpad * M + 255) / 256, 256, 0, stream>>>(Wq, Wk, Wv, Ws, bq, bk, bv, bs,
                                                         Wc, bc, fi, fo, Kpad);

    int tiles_m64 = M / 64, tiles_r32 = N / 32;
    long long thr = (long long)tiles_m64 * tiles_r32 * 32;
    k_gemm<<<(unsigned)((thr + 255) / 256), 256, 0, stream>>>(hb, Wc, bc, qkvs, Kpad, M,
                                                              tiles_m64, tiles_r32);

    k_zero<<<(N + 255) / 256, 256, 0, stream>>>(mseg, N);
    k_zero<<<(N + 255) / 256, 256, 0, stream>>>((unsigned*)den, N);
    long long naf = (long long)N * fo;
    k_zero<<<(unsigned)((naf + 255) / 256), 256, 0, stream>>>((unsigned*)agg, naf);

    k_scores<<<(unsigned)(((long long)E * 32 + 255) / 256), 256, 0, stream>>>(
        qkvs, src, dst, scores, mseg, fo);
    k_expden<<<(E + 255) / 256, 256, 0, stream>>>(scores, dst, mseg, aexp, den);
    long long eaf = (long long)E * fo;
    k_agg<<<(unsigned)((eaf + 255) / 256), 256, 0, stream>>>(qkvs, src, dst, aexp, den,
                                                             agg, fo);
    k_combine<<<(unsigned)((naf + 255) / 256), 256, 0, stream>>>(agg, qkvs, h, hb, fo);
  }

  // attention pooling
  k_zero<<<1, 256, 0, stream>>>(gm, G);
  k_zero<<<1, 256, 0, stream>>>((unsigned*)gden, G);
  k_zero<<<(G * 32 + 255) / 256, 256, 0, stream>>>((unsigned*)pooled, G * 32);
  k_gate<<<(N * 32 + 255) / 256, 256, 0, stream>>>(h, Wg, bg, batch, gate, gm);
  k_poolexp<<<(N + 255) / 256, 256, 0, stream>>>(gate, batch, gm, gden);
  k_pool<<<(N * 32 + 255) / 256, 256, 0, stream>>>(h, gate, batch, gden, pooled);
  k_final<<<2, 256, 0, stream>>>(pooled, Wfc, bfc, (float*)d_out);
}
